// MultiHeadAttention_54941221651184
// MI455X (gfx1250) — matrix operand order repaired and checked by reading, not compiled, not run on hardware
//
#include <hip/hip_runtime.h>
#include <hip/hip_bf16.h>

#define D_INN 1024
#define D_OUTT 1024
#define NHEAD 16
#define HDIM 64
#define BATCH 2
#define SEQ 2048
#define BT (BATCH * SEQ)
#define NEG_BIG (-3.0e38f)

typedef __attribute__((ext_vector_type(16))) _Float16 v16h;
typedef __attribute__((ext_vector_type(8)))  _Float16 v8h;
typedef __attribute__((ext_vector_type(2)))  _Float16 v2h;
typedef __attribute__((ext_vector_type(8)))  float    v8f;

#define WMMA_F16(a, b, c) \
  __builtin_amdgcn_wmma_f32_16x16x32_f16(false, (a), false, (b), (short)0, (c), false, false)

// A fragment, 16x32 f16, from row-major memory (ld in elements).
// Lane l: row m = l&15; halves 0..7 -> K = 8*(l>>4)+h, halves 8..15 -> K = 16+8*(l>>4)+(h-8).
static __device__ __forceinline__ v16h load_a16x32(const _Float16* base, int ld, int lane) {
  const int m = lane & 15, hi = lane >> 4;
  const _Float16* p = base + (size_t)m * ld + 8 * hi;
  v8h lo = *(const v8h*)p;
  v8h hv = *(const v8h*)(p + 16);
  return __builtin_shufflevector(lo, hv, 0,1,2,3,4,5,6,7,8,9,10,11,12,13,14,15);
}

// B fragment, 32x16 f16, from TRANSPOSED storage Bt[n][k] (ld in elements).
// Lane l: col n = l&15; halves 0..7 -> K = 8*(l>>4)+h, halves 8..15 -> K = 16+8*(l>>4)+(h-8).
static __device__ __forceinline__ v16h load_b32x16(const _Float16* baseT, int ld, int lane) {
  const int n = lane & 15, hi = lane >> 4;
  const _Float16* p = baseT + (size_t)n * ld + 8 * hi;
  v8h lo = *(const v8h*)p;
  v8h hv = *(const v8h*)(p + 16);
  return __builtin_shufflevector(lo, hv, 0,1,2,3,4,5,6,7,8,9,10,11,12,13,14,15);
}

// reductions across the 16 lanes holding one C-matrix row (wave32, groups of 16)
static __device__ __forceinline__ float red_max16(float v) {
  v = fmaxf(v, __shfl_xor(v, 1, 32));
  v = fmaxf(v, __shfl_xor(v, 2, 32));
  v = fmaxf(v, __shfl_xor(v, 4, 32));
  v = fmaxf(v, __shfl_xor(v, 8, 32));
  return v;
}
static __device__ __forceinline__ float red_sum16(float v) {
  v += __shfl_xor(v, 1, 32);
  v += __shfl_xor(v, 2, 32);
  v += __shfl_xor(v, 4, 32);
  v += __shfl_xor(v, 8, 32);
  return v;
}

// ---------------- pack kernels ----------------

__global__ void cvt_f32_f16_kernel(const float* __restrict__ in, _Float16* __restrict__ outp, int n) {
  int i = blockIdx.x * blockDim.x + threadIdx.x;
  if (i < n) outp[i] = (_Float16)in[i];
}

// Wt[o][i] = (f16) W[i][o]; LDS-tiled transpose, blockDim (32,8), grid (32,32,4)
__global__ void transpose_cvt_kernel(const float* __restrict__ Wq, const float* __restrict__ Wk,
                                     const float* __restrict__ Wv, const float* __restrict__ Wo,
                                     _Float16* __restrict__ WqT, _Float16* __restrict__ WkT,
                                     _Float16* __restrict__ WvT, _Float16* __restrict__ WoT) {
  __shared__ float tile[32][33];
  const float* src;
  _Float16* dst;
  switch (blockIdx.z) {
    case 0: src = Wq; dst = WqT; break;
    case 1: src = Wk; dst = WkT; break;
    case 2: src = Wv; dst = WvT; break;
    default: src = Wo; dst = WoT; break;
  }
  const int i0 = blockIdx.y * 32;  // input row (d_in)
  const int o0 = blockIdx.x * 32;  // input col (d_out)
  const int tx = threadIdx.x, ty = threadIdx.y;
#pragma unroll
  for (int k = 0; k < 32; k += 8)
    tile[ty + k][tx] = src[(size_t)(i0 + ty + k) * D_OUTT + o0 + tx];
  __syncthreads();
#pragma unroll
  for (int k = 0; k < 32; k += 8)
    dst[(size_t)(o0 + ty + k) * D_INN + i0 + tx] = (_Float16)tile[tx][ty + k];
}

// ---------------- QKV projection GEMM (software pipelined) ----------------
// C[4096 x 1024] = xh[4096 x 1024] * W ; W given transposed f16.
// Each wave: 16(M) x 64(N); block = 8 waves; grid (32, 16, 3).
// Q,K stored [b,h,t,e]; V stored transposed [b,h,e,t] for the P@V stage.
__global__ __launch_bounds__(256) void qkv_gemm_kernel(
    const _Float16* __restrict__ xh,
    const _Float16* __restrict__ WqT, const _Float16* __restrict__ WkT,
    const _Float16* __restrict__ WvT,
    _Float16* __restrict__ Qh, _Float16* __restrict__ Kh, _Float16* __restrict__ Vt) {
  const int lane = threadIdx.x & 31, wid = threadIdx.x >> 5;
  const int mbase = blockIdx.x * 128 + wid * 16;
  const int nbase = blockIdx.y * 64;
  const int z = blockIdx.z;
  const _Float16* WT = (z == 0) ? WqT : (z == 1) ? WkT : WvT;

  v8f zero = {};
  v8f acc[4];
  acc[0] = zero; acc[1] = zero; acc[2] = zero; acc[3] = zero;

  const _Float16* arow = xh + (size_t)mbase * D_INN;
  const _Float16* b0 = WT + (size_t)(nbase +  0) * D_INN;
  const _Float16* b1 = WT + (size_t)(nbase + 16) * D_INN;
  const _Float16* b2 = WT + (size_t)(nbase + 32) * D_INN;
  const _Float16* b3 = WT + (size_t)(nbase + 48) * D_INN;

  // prologue: fragments for kk = 0
  v16h ac  = load_a16x32(arow, D_INN, lane);
  v16h bc0 = load_b32x16(b0, D_INN, lane);
  v16h bc1 = load_b32x16(b1, D_INN, lane);
  v16h bc2 = load_b32x16(b2, D_INN, lane);
  v16h bc3 = load_b32x16(b3, D_INN, lane);

#pragma unroll 2
  for (int kk = 0; kk < D_INN; kk += 32) {
    // issue next iteration's loads first (branchless in-bounds wrap on last iter)
    const int kn = (kk + 32) & (D_INN - 1);
    v16h an  = load_a16x32(arow + kn, D_INN, lane);
    v16h bn0 = load_b32x16(b0 + kn, D_INN, lane);
    v16h bn1 = load_b32x16(b1 + kn, D_INN, lane);
    v16h bn2 = load_b32x16(b2 + kn, D_INN, lane);
    v16h bn3 = load_b32x16(b3 + kn, D_INN, lane);
    // compute on resident fragments while next loads are in flight
    acc[0] = WMMA_F16(ac, bc0, acc[0]);
    acc[1] = WMMA_F16(ac, bc1, acc[1]);
    acc[2] = WMMA_F16(ac, bc2, acc[2]);
    acc[3] = WMMA_F16(ac, bc3, acc[3]);
    ac = an; bc0 = bn0; bc1 = bn1; bc2 = bn2; bc3 = bn3;
  }

  const int nl = lane & 15, hi = lane >> 4;
  if (z == 2) {
    // V: rows of a C fragment are consecutive t in [b,h,e,t] layout -> b128 store
#pragma unroll
    for (int j = 0; j < 4; ++j) {
      const int col = nbase + j * 16 + nl;
      const int h = col >> 6, e = col & (HDIM - 1);
      const int row0 = mbase + 8 * hi;
      const int b = row0 >> 11, t0 = row0 & (SEQ - 1);
      v8h pk;
#pragma unroll
      for (int r = 0; r < 8; ++r) pk[r] = (_Float16)acc[j][r];
      *(v8h*)(Vt + ((size_t)((b * NHEAD + h) * HDIM + e)) * SEQ + t0) = pk;
    }
  } else {
    _Float16* dst = (z == 0) ? Qh : Kh;
#pragma unroll
    for (int j = 0; j < 4; ++j) {
      const int col = nbase + j * 16 + nl;
      const int h = col >> 6, e = col & (HDIM - 1);
#pragma unroll
      for (int r = 0; r < 8; ++r) {
        const int row = mbase + r + 8 * hi;          // 0..4095
        const int b = row >> 11, t = row & (SEQ - 1);
        dst[(((size_t)(b * NHEAD + h)) * SEQ + t) * HDIM + e] = (_Float16)acc[j][r];
      }
    }
  }
}

// ---------------- flash attention (pipelined over key chunks) ----------------
// One wave = one 16-query tile for one (b,h). Online softmax over 32-key chunks.
__global__ __launch_bounds__(128) void attn_fwd_kernel(
    const _Float16* __restrict__ Qh, const _Float16* __restrict__ Kh,
    const _Float16* __restrict__ Vt, const unsigned char* __restrict__ kmask,
    _Float16* __restrict__ ctx) {
  __shared__ _Float16 ldsP[4][16 * 32];
  const int lane = threadIdx.x & 31, wid = threadIdx.x >> 5;
  const int gw = blockIdx.x * 4 + wid;
  const int qt = gw & (SEQ / 16 - 1);   // 128 query tiles
  const int bh = gw >> 7;               // 0..31
  const int b = bh >> 4;
  const int h = bh & (NHEAD - 1);
  const int qbase = qt * 16;
  const int nl = lane & 15, hi = lane >> 4;

  const _Float16* qp = Qh + ((size_t)bh * SEQ + qbase) * HDIM;
  const v16h aq0 = load_a16x32(qp, HDIM, lane);       // hd 0..31
  const v16h aq1 = load_a16x32(qp + 32, HDIM, lane);  // hd 32..63

  v8f zero = {};
  v8f acc[4];
  acc[0] = zero; acc[1] = zero; acc[2] = zero; acc[3] = zero;
  float rmax[8], rsum[8];
#pragma unroll
  for (int r = 0; r < 8; ++r) { rmax[r] = NEG_BIG; rsum[r] = 0.0f; }

  _Float16* lp = &ldsP[wid][0];
  const _Float16* kp0 = Kh + (size_t)bh * SEQ * HDIM;
  const _Float16* vbase = Vt + (size_t)bh * HDIM * SEQ;
  const int nchunks = (qbase + 47) >> 5;   // causal: keys 0 .. qbase+15

  // prologue: K fragments for chunk 0
  v16h kf0 = load_b32x16(kp0, HDIM, lane);
  v16h kf1 = load_b32x16(kp0 + 32, HDIM, lane);
  v16h kf2 = load_b32x16(kp0 + 16 * HDIM, HDIM, lane);
  v16h kf3 = load_b32x16(kp0 + 16 * HDIM + 32, HDIM, lane);

  for (int c = 0; c < nchunks; ++c) {
    const int kbase = c * 32;

    // S = Q * K^T  (two 16x16 score fragments, K-dim = hd = 64)
    v8f s0 = zero, s1 = zero;
    s0 = WMMA_F16(aq0, kf0, s0);
    s0 = WMMA_F16(aq1, kf1, s0);
    s1 = WMMA_F16(aq0, kf2, s1);
    s1 = WMMA_F16(aq1, kf3, s1);

    // issue this chunk's V loads + next chunk's K loads now; they stream in
    // under the softmax VALU work and the P LDS round-trip below.
    const _Float16* vp = vbase + kbase;
    v16h vf0 = load_b32x16(vp +  0 * SEQ, SEQ, lane);
    v16h vf1 = load_b32x16(vp + 16 * SEQ, SEQ, lane);
    v16h vf2 = load_b32x16(vp + 32 * SEQ, SEQ, lane);
    v16h vf3 = load_b32x16(vp + 48 * SEQ, SEQ, lane);

    int kbn = kbase + 32;
    if (kbn > SEQ - 32) kbn = kbase;     // clamp: stay in bounds (data unused)
    const _Float16* kpn = kp0 + (size_t)kbn * HDIM;
    v16h kn0 = load_b32x16(kpn, HDIM, lane);
    v16h kn1 = load_b32x16(kpn + 32, HDIM, lane);
    v16h kn2 = load_b32x16(kpn + 16 * HDIM, HDIM, lane);
    v16h kn3 = load_b32x16(kpn + 16 * HDIM + 32, HDIM, lane);

    const int key0 = kbase + nl, key1 = key0 + 16;
    const bool pad0 = kmask[(size_t)b * SEQ + key0] != 0;
    const bool pad1 = kmask[(size_t)b * SEQ + key1] != 0;

#pragma unroll
    for (int r = 0; r < 8; ++r) {
      const int q = qbase + r + 8 * hi;
      float x0 = (key0 > q || pad0) ? NEG_BIG : s0[r] * 0.125f;  // 1/sqrt(64)
      float x1 = (key1 > q || pad1) ? NEG_BIG : s1[r] * 0.125f;
      const float mloc = red_max16(fmaxf(x0, x1));
      const float mnew = fmaxf(rmax[r], mloc);
      const float p0 = __expf(x0 - mnew);
      const float p1 = __expf(x1 - mnew);
      const float corr = __expf(rmax[r] - mnew);
      rmax[r] = mnew;
      rsum[r] = rsum[r] * corr + red_sum16(p0 + p1);
      acc[0][r] *= corr; acc[1][r] *= corr; acc[2][r] *= corr; acc[3][r] *= corr;
      // interleaved LDS layout: idx = m*32 + n*2 + j  -> packed 32-bit store
      v2h pk = { (_Float16)p0, (_Float16)p1 };
      *(v2h*)(lp + ((r + 8 * hi) * 32 + 2 * nl)) = pk;
    }

    // Reload P as an A fragment (16x32). Contiguous 16-half run, then de-interleave.
    const _Float16* plp = lp + nl * 32 + 16 * hi;
    v8h t0 = *(const v8h*)plp;
    v8h t1 = *(const v8h*)(plp + 8);
    v16h tt = __builtin_shufflevector(t0, t1, 0,1,2,3,4,5,6,7,8,9,10,11,12,13,14,15);
    v16h ap = __builtin_shufflevector(tt, tt, 0,2,4,6,8,10,12,14,1,3,5,7,9,11,13,15);

    // ctx += P * V   (V stored transposed [hd][t] -> contiguous B fragments)
    acc[0] = WMMA_F16(ap, vf0, acc[0]);
    acc[1] = WMMA_F16(ap, vf1, acc[1]);
    acc[2] = WMMA_F16(ap, vf2, acc[2]);
    acc[3] = WMMA_F16(ap, vf3, acc[3]);

    kf0 = kn0; kf1 = kn1; kf2 = kn2; kf3 = kn3;
  }

  // normalize and store ctx in [b, t, h*64+e] f16
#pragma unroll
  for (int r = 0; r < 8; ++r) {
    const float inv = 1.0f / rsum[r];
    const int row = qbase + r + 8 * hi;
    _Float16* op = ctx + ((size_t)(b * SEQ + row)) * D_OUTT + h * HDIM + nl;
    op[0]  = (_Float16)(acc[0][r] * inv);
    op[16] = (_Float16)(acc[1][r] * inv);
    op[32] = (_Float16)(acc[2][r] * inv);
    op[48] = (_Float16)(acc[3][r] * inv);
  }
}

// ---------------- output projection GEMM (+bias, f32 out, pipelined) ----------------
__global__ __launch_bounds__(256) void out_gemm_kernel(
    const _Float16* __restrict__ ctx, const _Float16* __restrict__ WoT,
    const float* __restrict__ bo, float* __restrict__ out) {
  const int lane = threadIdx.x & 31, wid = threadIdx.x >> 5;
  const int mbase = blockIdx.x * 128 + wid * 16;
  const int nbase = blockIdx.y * 64;

  v8f zero = {};
  v8f acc[4];
  acc[0] = zero; acc[1] = zero; acc[2] = zero; acc[3] = zero;

  const _Float16* arow = ctx + (size_t)mbase * D_OUTT;
  const _Float16* b0 = WoT + (size_t)(nbase +  0) * D_OUTT;
  const _Float16* b1 = WoT + (size_t)(nbase + 16) * D_OUTT;
  const _Float16* b2 = WoT + (size_t)(nbase + 32) * D_OUTT;
  const _Float16* b3 = WoT + (size_t)(nbase + 48) * D_OUTT;

  v16h ac  = load_a16x32(arow, D_OUTT, lane);
  v16h bc0 = load_b32x16(b0, D_OUTT, lane);
  v16h bc1 = load_b32x16(b1, D_OUTT, lane);
  v16h bc2 = load_b32x16(b2, D_OUTT, lane);
  v16h bc3 = load_b32x16(b3, D_OUTT, lane);

#pragma unroll 2
  for (int kk = 0; kk < D_OUTT; kk += 32) {
    const int kn = (kk + 32) & (D_OUTT - 1);
    v16h an  = load_a16x32(arow + kn, D_OUTT, lane);
    v16h bn0 = load_b32x16(b0 + kn, D_OUTT, lane);
    v16h bn1 = load_b32x16(b1 + kn, D_OUTT, lane);
    v16h bn2 = load_b32x16(b2 + kn, D_OUTT, lane);
    v16h bn3 = load_b32x16(b3 + kn, D_OUTT, lane);
    acc[0] = WMMA_F16(ac, bc0, acc[0]);
    acc[1] = WMMA_F16(ac, bc1, acc[1]);
    acc[2] = WMMA_F16(ac, bc2, acc[2]);
    acc[3] = WMMA_F16(ac, bc3, acc[3]);
    ac = an; bc0 = bn0; bc1 = bn1; bc2 = bn2; bc3 = bn3;
  }

  const int nl = lane & 15, hi = lane >> 4;
#pragma unroll
  for (int j = 0; j < 4; ++j) {
    const int col = nbase + j * 16 + nl;
    const float bias = bo[col];
#pragma unroll
    for (int r = 0; r < 8; ++r) {
      const int row = mbase + r + 8 * hi;
      out[(size_t)row * D_OUTT + col] = acc[j][r] + bias;
    }
  }
}

// ---------------- launch ----------------

extern "C" void kernel_launch(void* const* d_in, const int* in_sizes, int n_in,
                              void* d_out, int out_size, void* d_ws, size_t ws_size,
                              hipStream_t stream) {
  (void)in_sizes; (void)n_in; (void)out_size; (void)ws_size;
  const float* x  = (const float*)d_in[0];
  const float* Wq = (const float*)d_in[1];
  const float* Wk = (const float*)d_in[2];
  const float* Wv = (const float*)d_in[3];
  const float* Wo = (const float*)d_in[4];
  const float* bo = (const float*)d_in[5];
  const unsigned char* kmask = (const unsigned char*)d_in[6];
  float* out = (float*)d_out;

  char* w = (char*)d_ws;
  _Float16* xh  = (_Float16*)w; w += (size_t)BT * D_INN * sizeof(_Float16);        // 8 MB
  _Float16* WqT = (_Float16*)w; w += (size_t)D_INN * D_OUTT * sizeof(_Float16);    // 2 MB
  _Float16* WkT = (_Float16*)w; w += (size_t)D_INN * D_OUTT * sizeof(_Float16);
  _Float16* WvT = (_Float16*)w; w += (size_t)D_INN * D_OUTT * sizeof(_Float16);
  _Float16* WoT = (_Float16*)w; w += (size_t)D_INN * D_OUTT * sizeof(_Float16);
  _Float16* Qh  = (_Float16*)w; w += (size_t)BT * D_OUTT * sizeof(_Float16);       // 8 MB
  _Float16* Kh  = (_Float16*)w; w += (size_t)BT * D_OUTT * sizeof(_Float16);
  _Float16* Vt  = (_Float16*)w; w += (size_t)BT * D_OUTT * sizeof(_Float16);
  _Float16* ctx = (_Float16*)w; w += (size_t)BT * D_OUTT * sizeof(_Float16);

  cvt_f32_f16_kernel<<<(BT * D_INN) / 256, 256, 0, stream>>>(x, xh, BT * D_INN);
  transpose_cvt_kernel<<<dim3(32, 32, 4), dim3(32, 8), 0, stream>>>(
      Wq, Wk, Wv, Wo, WqT, WkT, WvT, WoT);
  qkv_gemm_kernel<<<dim3(BT / 128, D_OUTT / 64, 3), 256, 0, stream>>>(
      xh, WqT, WkT, WvT, Qh, Kh, Vt);
  attn_fwd_kernel<<<(BATCH * NHEAD * (SEQ / 16)) / 4, 128, 0, stream>>>(
      Qh, Kh, Vt, kmask, ctx);
  out_gemm_kernel<<<dim3(BT / 128, D_OUTT / 64), 256, 0, stream>>>(ctx, WoT, bo, out);
}
